// MixtralDecoderLayer_55147380081303
// MI455X (gfx1250) — matrix operand order repaired and checked by reading, not compiled, not run on hardware
//
#include <hip/hip_runtime.h>
#include <stdint.h>

// ---------------- problem constants ----------------
#define T_TOK 2048
#define HDIM  2048
#define NHEAD 32
#define NKVH  8
#define HD    64
#define FFN   4096
#define NEXP  8
#define QKVN  3072   // (NH + 2*NKV) * HD

typedef unsigned short u16;
typedef __attribute__((ext_vector_type(4)))  unsigned int u32x4;
typedef __attribute__((ext_vector_type(4)))  int          i32x4;
typedef __attribute__((ext_vector_type(16))) __bf16       v16bf;
typedef __attribute__((ext_vector_type(8)))  float        v8f;

union ABu { v16bf v; u32x4 q[2]; };

__device__ __forceinline__ u16 f2bf(float f) {
  unsigned u = __float_as_uint(f);
  unsigned r = (u + 0x7fffu + ((u >> 16) & 1u)) >> 16;
  return (u16)r;
}
__device__ __forceinline__ float bf2f(u16 h) {
  return __uint_as_float(((unsigned)h) << 16);
}
__device__ __forceinline__ v8f vzero8() {
  v8f z;
#pragma unroll
  for (int i = 0; i < 8; ++i) z[i] = 0.0f;
  return z;
}
#define WMMA_BF16(A, B, C) \
  __builtin_amdgcn_wmma_f32_16x16x32_bf16(false, (A), false, (B), (short)0, (C), false, false)

// ---------------- CDNA5 async global->LDS copy (ASYNCcnt path) ----------------
#if defined(__has_builtin)
#if __has_builtin(__builtin_amdgcn_global_load_async_to_lds_b128)
#define USE_ASYNC_LDS 1
#endif
#endif

__device__ __forceinline__ void copy16_g2lds(u16* lds_dst, const u16* gsrc) {
#ifdef USE_ASYNC_LDS
  __builtin_amdgcn_global_load_async_to_lds_b128(
      (__attribute__((address_space(1))) i32x4*)(uintptr_t)(const void*)gsrc,
      (__attribute__((address_space(3))) i32x4*)lds_dst,
      0, 0);
#else
  *(u32x4*)lds_dst = *(const u32x4*)gsrc;
#endif
}
__device__ __forceinline__ void async_wait0() {
#ifdef USE_ASYNC_LDS
#if __has_builtin(__builtin_amdgcn_s_wait_asynccnt)
  __builtin_amdgcn_s_wait_asynccnt(0);
#else
  asm volatile("s_wait_asynccnt 0" ::: "memory");
#endif
#endif
}

// ---------------- RMSNorm (fp32 in -> bf16 out) ----------------
__global__ __launch_bounds__(256) void rmsnorm_kernel(
    const float* __restrict__ X, const float* __restrict__ W, u16* __restrict__ Y) {
  __shared__ float red[256];
  const int t = blockIdx.x, tid = threadIdx.x;
  const float* x = X + (size_t)t * HDIM;
  float s = 0.f;
  for (int c = tid; c < HDIM; c += 256) { float v = x[c]; s += v * v; }
  red[tid] = s; __syncthreads();
  for (int st = 128; st > 0; st >>= 1) {
    if (tid < st) red[tid] += red[tid + st];
    __syncthreads();
  }
  const float inv = rsqrtf(red[0] * (1.0f / HDIM) + 1e-5f);
  for (int c = tid; c < HDIM; c += 256)
    Y[(size_t)t * HDIM + c] = f2bf(x[c] * inv * W[c]);
}

// ---------------- generic GEMM: C[M,N] = A(bf16)[M,K] x B(f32->bf16)[K,N] (+addend) ----------------
// block tile 128x128, K-step 32, 8 waves (2x4), wave tile 64x32, double-buffered LDS
__global__ __launch_bounds__(256) void gemm_kernel(
    const u16* __restrict__ A, const float* __restrict__ B,
    float* __restrict__ C, const float* __restrict__ addend,
    int M, int N, int K) {
  __shared__ __align__(16) u16 lsA[2][128 * 32];  // [m][k]
  __shared__ __align__(16) u16 lsB[2][128 * 32];  // [n][k]
  const int tid = threadIdx.x, lane = tid & 31, wave = tid >> 5;
  const int l16 = lane & 15, lhi = lane >> 4;
  const int wm = wave >> 2, wn = wave & 3;
  const int bm0 = blockIdx.y * 128, bn0 = blockIdx.x * 128;
  v8f acc[4][2];
#pragma unroll
  for (int mi = 0; mi < 4; ++mi)
#pragma unroll
    for (int ni = 0; ni < 2; ++ni) acc[mi][ni] = vzero8();

  auto load_tiles = [&](int buf, int k0) {
#pragma unroll
    for (int i = 0; i < 2; ++i) {  // A tile: 512 16B chunks, async to LDS
      int idx = tid + i * 256;
      int row = idx >> 2, c8 = (idx & 3) * 8;
      copy16_g2lds(&lsA[buf][row * 32 + c8],
                   A + (size_t)(bm0 + row) * K + k0 + c8);
    }
    // B tile: f32 coalesced over n, store bf16 transposed [n][k]
    int n = tid & 127;
    int kg = (tid >> 7) * 16;
    const float* bp = B + (size_t)(k0 + kg) * N + bn0 + n;
    unsigned* dst = (unsigned*)(&lsB[buf][n * 32 + kg]);
#pragma unroll
    for (int i = 0; i < 16; i += 2) {
      float v0 = bp[(size_t)i * N], v1 = bp[(size_t)(i + 1) * N];
      dst[i >> 1] = (unsigned)f2bf(v0) | ((unsigned)f2bf(v1) << 16);
    }
  };

  load_tiles(0, 0);
  int cur = 0;
  for (int k0 = 0; k0 < K; k0 += 32) {
    async_wait0();
    __syncthreads();
    if (k0 + 32 < K) load_tiles(cur ^ 1, k0 + 32);
    ABu af[4], bf[2];
#pragma unroll
    for (int mi = 0; mi < 4; ++mi) {
      const u16* p = &lsA[cur][(wm * 64 + mi * 16 + l16) * 32 + lhi * 8];
      af[mi].q[0] = *(const u32x4*)p;
      af[mi].q[1] = *(const u32x4*)(p + 16);
    }
#pragma unroll
    for (int ni = 0; ni < 2; ++ni) {
      const u16* p = &lsB[cur][(wn * 32 + ni * 16 + l16) * 32 + lhi * 8];
      bf[ni].q[0] = *(const u32x4*)p;
      bf[ni].q[1] = *(const u32x4*)(p + 16);
    }
#pragma unroll
    for (int mi = 0; mi < 4; ++mi)
#pragma unroll
      for (int ni = 0; ni < 2; ++ni)
        acc[mi][ni] = WMMA_BF16(af[mi].v, bf[ni].v, acc[mi][ni]);
    cur ^= 1;
  }
  if (addend) {
#pragma unroll
    for (int mi = 0; mi < 4; ++mi)
#pragma unroll
      for (int ni = 0; ni < 2; ++ni) {
        int gn = bn0 + wn * 32 + ni * 16 + l16;
#pragma unroll
        for (int r = 0; r < 8; ++r) {
          int gm = bm0 + wm * 64 + mi * 16 + r + 8 * lhi;
          C[(size_t)gm * N + gn] = acc[mi][ni][r] + addend[(size_t)gm * N + gn];
        }
      }
  } else {
#pragma unroll
    for (int mi = 0; mi < 4; ++mi)
#pragma unroll
      for (int ni = 0; ni < 2; ++ni) {
        int gn = bn0 + wn * 32 + ni * 16 + l16;
#pragma unroll
        for (int r = 0; r < 8; ++r) {
          int gm = bm0 + wm * 64 + mi * 16 + r + 8 * lhi;
          C[(size_t)gm * N + gn] = acc[mi][ni][r];
        }
      }
  }
}

// ---------------- RoPE + bf16 split of fused QKV ----------------
__global__ __launch_bounds__(256) void rope_kernel(
    const float* __restrict__ QKV, const int* __restrict__ pos,
    u16* __restrict__ Qo, u16* __restrict__ Ko, u16* __restrict__ Vo) {
  const int t = blockIdx.x, tid = threadIdx.x;
  const float fp = (float)pos[t];
  const float* base = QKV + (size_t)t * QKVN;
  for (int idx = tid; idx < NHEAD * 32; idx += 256) {
    int head = idx >> 5, d = idx & 31;
    float inv = __expf(-((float)(2 * d) / 64.f) * 9.2103403720f);  // ln(10000)
    float ang = fp * inv, sn, cs;
    __sincosf(ang, &sn, &cs);
    float x1 = base[head * 64 + d], x2 = base[head * 64 + d + 32];
    Qo[(size_t)t * (NHEAD * HD) + head * 64 + d]      = f2bf(x1 * cs - x2 * sn);
    Qo[(size_t)t * (NHEAD * HD) + head * 64 + d + 32] = f2bf(x2 * cs + x1 * sn);
  }
  if (tid < NKVH * 32) {
    int head = tid >> 5, d = tid & 31;
    float inv = __expf(-((float)(2 * d) / 64.f) * 9.2103403720f);
    float ang = fp * inv, sn, cs;
    __sincosf(ang, &sn, &cs);
    const float* kbp = base + NHEAD * HD;
    float x1 = kbp[head * 64 + d], x2 = kbp[head * 64 + d + 32];
    Ko[(size_t)t * (NKVH * HD) + head * 64 + d]      = f2bf(x1 * cs - x2 * sn);
    Ko[(size_t)t * (NKVH * HD) + head * 64 + d + 32] = f2bf(x2 * cs + x1 * sn);
  }
  for (int c = tid; c < NKVH * HD; c += 256)
    Vo[(size_t)t * (NKVH * HD) + c] = f2bf(base[(NHEAD + NKVH) * HD + c]);
}

// ---------------- flash attention (causal, GQA 4:1), bf16 WMMA ----------------
// grid (T/128, NH); 8 waves, wave handles 16 q rows; key blocks of 64
__global__ __launch_bounds__(256) void attn_kernel(
    const u16* __restrict__ Q, const u16* __restrict__ Kb,
    const u16* __restrict__ Vb, u16* __restrict__ O) {
  __shared__ __align__(16) u16 lsK[64 * 64];      // [key][d]
  __shared__ __align__(16) u16 lsV[64 * 64];      // [d][key] transposed
  __shared__ __align__(16) u16 lsP[8][16 * 32];   // per-wave prob bounce
  const int tid = threadIdx.x, lane = tid & 31, wave = tid >> 5;
  const int l16 = lane & 15, lhi = lane >> 4;
  const int h = blockIdx.y;
  const int kvh = h >> 2;
  const int qb = blockIdx.x;
  const int qrow0 = qb * 128 + wave * 16;
  const float scale = 0.125f;  // 1/sqrt(64)

  ABu qf[2];
  {
    const u16* qp = Q + (size_t)(qrow0 + l16) * (NHEAD * HD) + h * 64;
#pragma unroll
    for (int f = 0; f < 2; ++f) {
      qf[f].q[0] = *(const u32x4*)(qp + f * 32 + lhi * 8);
      qf[f].q[1] = *(const u32x4*)(qp + f * 32 + 16 + lhi * 8);
    }
  }
  v8f o[4];
#pragma unroll
  for (int df = 0; df < 4; ++df) o[df] = vzero8();
  float rmax[8], rsum[8];
#pragma unroll
  for (int r = 0; r < 8; ++r) { rmax[r] = -1e30f; rsum[r] = 0.f; }

  const int nkb = 2 * qb + 2;
  for (int kb = 0; kb < nkb; ++kb) {
    const int kbase = kb * 64;
    __syncthreads();
#pragma unroll
    for (int i = 0; i < 2; ++i) {  // K tile async copy; V tile manual transpose
      int c = tid * 2 + i;         // 0..511 chunks of 8 halfs
      int key = c >> 3, d8 = (c & 7) * 8;
      copy16_g2lds(&lsK[key * 64 + d8],
                   Kb + (size_t)(kbase + key) * (NKVH * HD) + kvh * 64 + d8);
      union { u32x4 v; u16 s[8]; } uv;
      uv.v = *(const u32x4*)(Vb + (size_t)(kbase + key) * (NKVH * HD) + kvh * 64 + d8);
#pragma unroll
      for (int j = 0; j < 8; ++j) lsV[(d8 + j) * 64 + key] = uv.s[j];
    }
    async_wait0();
    __syncthreads();
    if (kbase > qrow0 + 15) continue;  // fully masked for this wave

    for (int kc = 0; kc < 2; ++kc) {   // 32-key chunks
      const int kcb = kbase + kc * 32;
      if (kcb > qrow0 + 15) break;
      v8f sfr[2];
#pragma unroll
      for (int sub = 0; sub < 2; ++sub) {
        v8f s = vzero8();
#pragma unroll
        for (int dk = 0; dk < 2; ++dk) {
          ABu bk;
          const u16* p2 = lsK + (kc * 32 + sub * 16 + l16) * 64 + dk * 32 + lhi * 8;
          bk.q[0] = *(const u32x4*)p2;
          bk.q[1] = *(const u32x4*)(p2 + 16);
          s = WMMA_BF16(qf[dk].v, bk.v, s);
        }
        sfr[sub] = s;
      }
#pragma unroll
      for (int r = 0; r < 8; ++r) {  // online softmax over these 32 keys
        int gm = qrow0 + r + 8 * lhi;
        float sv0 = sfr[0][r] * scale;
        float sv1 = sfr[1][r] * scale;
        if (kcb + l16 > gm)      sv0 = -1e30f;
        if (kcb + 16 + l16 > gm) sv1 = -1e30f;
        float fm = fmaxf(rmax[r], fmaxf(sv0, sv1));
        for (int ofs = 1; ofs < 16; ofs <<= 1) fm = fmaxf(fm, __shfl_xor(fm, ofs, 32));
        float corr = __expf(rmax[r] - fm);
        float p0 = __expf(sv0 - fm), p1 = __expf(sv1 - fm);
        float ps = p0 + p1;
        for (int ofs = 1; ofs < 16; ofs <<= 1) ps += __shfl_xor(ps, ofs, 32);
        rsum[r] = rsum[r] * corr + ps;
        rmax[r] = fm;
#pragma unroll
        for (int df = 0; df < 4; ++df) o[df][r] *= corr;
        lsP[wave][(r + 8 * lhi) * 32 + l16]      = f2bf(p0);
        lsP[wave][(r + 8 * lhi) * 32 + 16 + l16] = f2bf(p1);
      }
      asm volatile("s_wait_dscnt 0" ::: "memory");  // wave-local LDS bounce
      ABu pf;
      {
        const u16* pp = &lsP[wave][0] + l16 * 32 + lhi * 8;
        pf.q[0] = *(const u32x4*)pp;
        pf.q[1] = *(const u32x4*)(pp + 16);
      }
#pragma unroll
      for (int df = 0; df < 4; ++df) {  // P x V
        ABu vf;
        const u16* vp = lsV + (df * 16 + l16) * 64 + kc * 32 + lhi * 8;
        vf.q[0] = *(const u32x4*)vp;
        vf.q[1] = *(const u32x4*)(vp + 16);
        o[df] = WMMA_BF16(pf.v, vf.v, o[df]);
      }
    }
  }
#pragma unroll
  for (int r = 0; r < 8; ++r) {
    float inv = 1.f / rsum[r];
    int gm = qrow0 + r + 8 * lhi;
#pragma unroll
    for (int df = 0; df < 4; ++df)
      O[(size_t)gm * (NHEAD * HD) + h * 64 + df * 16 + l16] = f2bf(o[df][r] * inv);
  }
}

// ---------------- router: logits, softmax, top-2 ----------------
__global__ __launch_bounds__(256) void router_kernel(
    const u16* __restrict__ X, const float* __restrict__ Gw,
    float* __restrict__ tw, int* __restrict__ tids) {
  __shared__ float red[256];
  __shared__ float logits[NEXP];
  const int t = blockIdx.x, tid = threadIdx.x;
  float p[NEXP];
#pragma unroll
  for (int e = 0; e < NEXP; ++e) p[e] = 0.f;
  for (int c = tid; c < HDIM; c += 256) {
    float xv = bf2f(X[(size_t)t * HDIM + c]);
    const float* g = Gw + (size_t)c * NEXP;
#pragma unroll
    for (int e = 0; e < NEXP; ++e) p[e] += xv * g[e];
  }
  for (int e = 0; e < NEXP; ++e) {
    red[tid] = p[e]; __syncthreads();
    for (int st = 128; st > 0; st >>= 1) {
      if (tid < st) red[tid] += red[tid + st];
      __syncthreads();
    }
    if (tid == 0) logits[e] = red[0];
    __syncthreads();
  }
  if (tid == 0) {
    float mx = -1e30f;
    for (int e = 0; e < NEXP; ++e) mx = fmaxf(mx, logits[e]);
    float pr[NEXP], se = 0.f;
    for (int e = 0; e < NEXP; ++e) { pr[e] = __expf(logits[e] - mx); se += pr[e]; }
    for (int e = 0; e < NEXP; ++e) pr[e] /= se;
    int i0 = 0;
    for (int e = 1; e < NEXP; ++e) if (pr[e] > pr[i0]) i0 = e;
    int i1 = -1;
    for (int e = 0; e < NEXP; ++e) if (e != i0 && (i1 < 0 || pr[e] > pr[i1])) i1 = e;
    float w0 = pr[i0], w1 = pr[i1], s = w0 + w1;
    tw[t * 2] = w0 / s; tw[t * 2 + 1] = w1 / s;
    tids[t * 2] = i0; tids[t * 2 + 1] = i1;
  }
}

// ---------------- deterministic per-expert compaction (ballot scan) ----------------
__global__ __launch_bounds__(256) void compact_kernel(
    const int* __restrict__ ids, const float* __restrict__ tw,
    int* __restrict__ lists, int* __restrict__ tgt,
    float* __restrict__ scl, int* __restrict__ counts) {
  const int e = blockIdx.x, tid = threadIdx.x, lane = tid & 31, wv = tid >> 5;
  __shared__ int sbase;
  __shared__ int wbase[8];
  if (tid == 0) sbase = 0;
  __syncthreads();
  for (int t0 = 0; t0 < T_TOK; t0 += 256) {
    int t = t0 + tid;
    int slot = -1; float w = 0.f;
    int i0 = ids[t * 2], i1 = ids[t * 2 + 1];
    if (i0 == e) { slot = 0; w = tw[t * 2]; }
    else if (i1 == e) { slot = 1; w = tw[t * 2 + 1]; }
    bool pred = slot >= 0;
    unsigned mask = (unsigned)__ballot(pred);
    int rank = __popc(mask & ((1u << lane) - 1u));
    if (lane == 0) wbase[wv] = __popc(mask);
    __syncthreads();
    if (tid == 0) {
      int a = sbase;
      for (int k = 0; k < 8; ++k) { int c = wbase[k]; wbase[k] = a; a += c; }
      sbase = a;
    }
    __syncthreads();
    if (pred) {
      int posn = wbase[wv] + rank;
      lists[e * T_TOK + posn] = t;
      tgt[e * T_TOK + posn] = t * 2 + slot;
      scl[e * T_TOK + posn] = w;
    }
    __syncthreads();
  }
  if (tid == 0) counts[e] = sbase;
}

__global__ void offsets_kernel(const int* __restrict__ counts, int* __restrict__ off) {
  if (threadIdx.x == 0 && blockIdx.x == 0) {
    int a = 0;
    for (int e = 0; e < NEXP; ++e) { off[e] = a; a += counts[e]; }
    off[NEXP] = a;
  }
}

// ---------------- MoE GEMM1: h = silu(x@w1[e]) * (x@w3[e]), gathered rows ----------------
// block tile 64x128, 8 waves (2x4), wave tile 32x32, dual accumulators, double-buffered
__global__ __launch_bounds__(256) void moe_gemm1_kernel(
    const u16* __restrict__ X, const float* __restrict__ W1, const float* __restrict__ W3,
    const int* __restrict__ lists, const int* __restrict__ counts,
    const int* __restrict__ hoff, u16* __restrict__ Hout) {
  const int e = blockIdx.z;
  const int cnt = counts[e];
  const int bm0 = blockIdx.y * 64;
  if (bm0 >= cnt) return;
  const int bn0 = blockIdx.x * 128;
  __shared__ __align__(16) u16 lsA[2][64 * 32];
  __shared__ __align__(16) u16 lsB1[2][128 * 32];
  __shared__ __align__(16) u16 lsB3[2][128 * 32];
  const int tid = threadIdx.x, lane = tid & 31, wave = tid >> 5;
  const int l16 = lane & 15, lhi = lane >> 4;
  const int wm = wave >> 2, wn = wave & 3;
  const float* B1 = W1 + (size_t)e * HDIM * FFN;
  const float* B3 = W3 + (size_t)e * HDIM * FFN;
  const int* lst = lists + e * T_TOK;
  v8f ac1[2][2], ac3[2][2];
#pragma unroll
  for (int mi = 0; mi < 2; ++mi)
#pragma unroll
    for (int ni = 0; ni < 2; ++ni) { ac1[mi][ni] = vzero8(); ac3[mi][ni] = vzero8(); }

  auto load_tiles = [&](int buf, int k0) {
    {
      int row = tid >> 2, c8 = (tid & 3) * 8;
      int rr = bm0 + row; if (rr >= cnt) rr = cnt - 1;
      int tok = lst[rr];
      copy16_g2lds(&lsA[buf][row * 32 + c8],
                   X + (size_t)tok * HDIM + k0 + c8);
    }
    int n = tid & 127, kg = (tid >> 7) * 16;
    const float* bp1 = B1 + (size_t)(k0 + kg) * FFN + bn0 + n;
    const float* bp3 = B3 + (size_t)(k0 + kg) * FFN + bn0 + n;
    unsigned* d1 = (unsigned*)(&lsB1[buf][n * 32 + kg]);
    unsigned* d3 = (unsigned*)(&lsB3[buf][n * 32 + kg]);
#pragma unroll
    for (int i = 0; i < 16; i += 2) {
      d1[i >> 1] = (unsigned)f2bf(bp1[(size_t)i * FFN]) |
                   ((unsigned)f2bf(bp1[(size_t)(i + 1) * FFN]) << 16);
      d3[i >> 1] = (unsigned)f2bf(bp3[(size_t)i * FFN]) |
                   ((unsigned)f2bf(bp3[(size_t)(i + 1) * FFN]) << 16);
    }
  };

  load_tiles(0, 0);
  int cur = 0;
  for (int k0 = 0; k0 < HDIM; k0 += 32) {
    async_wait0();
    __syncthreads();
    if (k0 + 32 < HDIM) load_tiles(cur ^ 1, k0 + 32);
    ABu af[2], b1f[2], b3f[2];
#pragma unroll
    for (int mi = 0; mi < 2; ++mi) {
      const u16* p = &lsA[cur][(wm * 32 + mi * 16 + l16) * 32 + lhi * 8];
      af[mi].q[0] = *(const u32x4*)p;
      af[mi].q[1] = *(const u32x4*)(p + 16);
    }
#pragma unroll
    for (int ni = 0; ni < 2; ++ni) {
      const u16* p1 = &lsB1[cur][(wn * 32 + ni * 16 + l16) * 32 + lhi * 8];
      b1f[ni].q[0] = *(const u32x4*)p1;
      b1f[ni].q[1] = *(const u32x4*)(p1 + 16);
      const u16* p3 = &lsB3[cur][(wn * 32 + ni * 16 + l16) * 32 + lhi * 8];
      b3f[ni].q[0] = *(const u32x4*)p3;
      b3f[ni].q[1] = *(const u32x4*)(p3 + 16);
    }
#pragma unroll
    for (int mi = 0; mi < 2; ++mi)
#pragma unroll
      for (int ni = 0; ni < 2; ++ni) {
        ac1[mi][ni] = WMMA_BF16(af[mi].v, b1f[ni].v, ac1[mi][ni]);
        ac3[mi][ni] = WMMA_BF16(af[mi].v, b3f[ni].v, ac3[mi][ni]);
      }
    cur ^= 1;
  }
  const int h0 = hoff[e];
#pragma unroll
  for (int mi = 0; mi < 2; ++mi)
#pragma unroll
    for (int ni = 0; ni < 2; ++ni) {
      int gn = bn0 + wn * 32 + ni * 16 + l16;
#pragma unroll
      for (int r = 0; r < 8; ++r) {
        int gm = bm0 + wm * 32 + mi * 16 + r + 8 * lhi;
        if (gm < cnt) {
          float a = ac1[mi][ni][r];
          float hv = a * (1.f / (1.f + __expf(-a))) * ac3[mi][ni][r];
          Hout[((size_t)(h0 + gm)) * FFN + gn] = f2bf(hv);
        }
      }
    }
}

// ---------------- MoE GEMM2: partial[tgt] = (h @ w2[e]) * scale ----------------
__global__ __launch_bounds__(256) void moe_gemm2_kernel(
    const u16* __restrict__ Hbuf, const float* __restrict__ W2,
    const int* __restrict__ tgt, const float* __restrict__ scl,
    const int* __restrict__ counts, const int* __restrict__ hoff,
    float* __restrict__ Pout) {
  const int e = blockIdx.z;
  const int cnt = counts[e];
  const int bm0 = blockIdx.y * 128;
  if (bm0 >= cnt) return;
  const int bn0 = blockIdx.x * 128;
  __shared__ __align__(16) u16 lsA[2][128 * 32];
  __shared__ __align__(16) u16 lsB[2][128 * 32];
  const int tid = threadIdx.x, lane = tid & 31, wave = tid >> 5;
  const int l16 = lane & 15, lhi = lane >> 4;
  const int wm = wave >> 2, wn = wave & 3;
  const u16* A = Hbuf + (size_t)hoff[e] * FFN;
  const float* B = W2 + (size_t)e * FFN * HDIM;
  v8f acc[4][2];
#pragma unroll
  for (int mi = 0; mi < 4; ++mi)
#pragma unroll
    for (int ni = 0; ni < 2; ++ni) acc[mi][ni] = vzero8();

  auto load_tiles = [&](int buf, int k0) {
#pragma unroll
    for (int i = 0; i < 2; ++i) {
      int idx = tid + i * 256;
      int row = idx >> 2, c8 = (idx & 3) * 8;
      int rr = bm0 + row; if (rr >= cnt) rr = cnt - 1;
      copy16_g2lds(&lsA[buf][row * 32 + c8],
                   A + (size_t)rr * FFN + k0 + c8);
    }
    int n = tid & 127, kg = (tid >> 7) * 16;
    const float* bp = B + (size_t)(k0 + kg) * HDIM + bn0 + n;
    unsigned* dst = (unsigned*)(&lsB[buf][n * 32 + kg]);
#pragma unroll
    for (int i = 0; i < 16; i += 2) {
      dst[i >> 1] = (unsigned)f2bf(bp[(size_t)i * HDIM]) |
                    ((unsigned)f2bf(bp[(size_t)(i + 1) * HDIM]) << 16);
    }
  };

  load_tiles(0, 0);
  int cur = 0;
  for (int k0 = 0; k0 < FFN; k0 += 32) {
    async_wait0();
    __syncthreads();
    if (k0 + 32 < FFN) load_tiles(cur ^ 1, k0 + 32);
    ABu af[4], bf[2];
#pragma unroll
    for (int mi = 0; mi < 4; ++mi) {
      const u16* p = &lsA[cur][(wm * 64 + mi * 16 + l16) * 32 + lhi * 8];
      af[mi].q[0] = *(const u32x4*)p;
      af[mi].q[1] = *(const u32x4*)(p + 16);
    }
#pragma unroll
    for (int ni = 0; ni < 2; ++ni) {
      const u16* p = &lsB[cur][(wn * 32 + ni * 16 + l16) * 32 + lhi * 8];
      bf[ni].q[0] = *(const u32x4*)p;
      bf[ni].q[1] = *(const u32x4*)(p + 16);
    }
#pragma unroll
    for (int mi = 0; mi < 4; ++mi)
#pragma unroll
      for (int ni = 0; ni < 2; ++ni)
        acc[mi][ni] = WMMA_BF16(af[mi].v, bf[ni].v, acc[mi][ni]);
    cur ^= 1;
  }
#pragma unroll
  for (int mi = 0; mi < 4; ++mi)
#pragma unroll
    for (int ni = 0; ni < 2; ++ni) {
      int gn = bn0 + wn * 32 + ni * 16 + l16;
#pragma unroll
      for (int r = 0; r < 8; ++r) {
        int gm = bm0 + wm * 64 + mi * 16 + r + 8 * lhi;
        if (gm < cnt) {
          int tr = tgt[e * T_TOK + gm];
          float sc = scl[e * T_TOK + gm];
          Pout[(size_t)tr * HDIM + gn] = acc[mi][ni][r] * sc;
        }
      }
    }
}

// ---------------- sum the two expert slots ----------------
__global__ __launch_bounds__(256) void finaladd_kernel(
    const float* __restrict__ P, float* __restrict__ out) {
  size_t i = (size_t)blockIdx.x * 256 + threadIdx.x;
  size_t t = i >> 11, c = i & 2047;
  out[i] = P[(t * 2) * HDIM + c] + P[(t * 2 + 1) * HDIM + c];
}

// ---------------- host orchestration ----------------
extern "C" void kernel_launch(void* const* d_in, const int* in_sizes, int n_in,
                              void* d_out, int out_size, void* d_ws, size_t ws_size,
                              hipStream_t stream) {
  const float* hidden = (const float*)d_in[0];
  const int*   pos    = (const int*)d_in[1];
  const float* w_qkv  = (const float*)d_in[2];
  const float* w_o    = (const float*)d_in[3];
  const float* gate_w = (const float*)d_in[4];
  const float* w1     = (const float*)d_in[5];
  const float* w2     = (const float*)d_in[6];
  const float* w3     = (const float*)d_in[7];
  const float* ln1    = (const float*)d_in[8];
  const float* ln2    = (const float*)d_in[9];

  float* out_final = (float*)d_out;
  float* residual  = out_final + (size_t)T_TOK * HDIM;

  char* ws = (char*)d_ws;
  size_t off = 0;
  auto alloc = [&](size_t b) { size_t o = off; off += (b + 255) & ~(size_t)255; return o; };
  u16*   x1     = (u16*)(ws + alloc((size_t)T_TOK * HDIM * 2));
  float* qkv    = (float*)(ws + alloc((size_t)T_TOK * QKVN * 4));
  u16*   qb     = (u16*)(ws + alloc((size_t)T_TOK * NHEAD * HD * 2));
  u16*   kb     = (u16*)(ws + alloc((size_t)T_TOK * NKVH * HD * 2));
  u16*   vb     = (u16*)(ws + alloc((size_t)T_TOK * NKVH * HD * 2));
  u16*   attnb  = (u16*)(ws + alloc((size_t)T_TOK * NHEAD * HD * 2));
  u16*   x2     = (u16*)(ws + alloc((size_t)T_TOK * HDIM * 2));
  float* tw     = (float*)(ws + alloc((size_t)T_TOK * 2 * 4));
  int*   tids   = (int*)(ws + alloc((size_t)T_TOK * 2 * 4));
  int*   lists  = (int*)(ws + alloc((size_t)NEXP * T_TOK * 4));
  int*   tgt    = (int*)(ws + alloc((size_t)NEXP * T_TOK * 4));
  float* scl    = (float*)(ws + alloc((size_t)NEXP * T_TOK * 4));
  int*   counts = (int*)(ws + alloc(64));
  int*   hoff   = (int*)(ws + alloc(64));
  u16*   hbuf   = (u16*)(ws + alloc((size_t)2 * T_TOK * FFN * 2));
  float* part   = (float*)(ws + alloc((size_t)2 * T_TOK * HDIM * 4));
  (void)in_sizes; (void)n_in; (void)out_size; (void)ws_size;

  rmsnorm_kernel<<<T_TOK, 256, 0, stream>>>(hidden, ln1, x1);
  gemm_kernel<<<dim3(QKVN / 128, T_TOK / 128), 256, 0, stream>>>(
      x1, w_qkv, qkv, nullptr, T_TOK, QKVN, HDIM);
  rope_kernel<<<T_TOK, 256, 0, stream>>>(qkv, pos, qb, kb, vb);
  attn_kernel<<<dim3(T_TOK / 128, NHEAD), 256, 0, stream>>>(qb, kb, vb, attnb);
  gemm_kernel<<<dim3(HDIM / 128, T_TOK / 128), 256, 0, stream>>>(
      attnb, w_o, residual, hidden, T_TOK, HDIM, HDIM);
  rmsnorm_kernel<<<T_TOK, 256, 0, stream>>>(residual, ln2, x2);
  router_kernel<<<T_TOK, 256, 0, stream>>>(x2, gate_w, tw, tids);
  compact_kernel<<<NEXP, 256, 0, stream>>>(tids, tw, lists, tgt, scl, counts);
  offsets_kernel<<<1, 32, 0, stream>>>(counts, hoff);
  moe_gemm1_kernel<<<dim3(FFN / 128, T_TOK / 64, NEXP), 256, 0, stream>>>(
      x2, w1, w3, lists, counts, hoff, hbuf);
  moe_gemm2_kernel<<<dim3(HDIM / 128, T_TOK / 128, NEXP), 256, 0, stream>>>(
      hbuf, w2, tgt, scl, counts, hoff, part);
  finaladd_kernel<<<(T_TOK * HDIM) / 256, 256, 0, stream>>>(part, out_final);
}
